// GraphResBlock_69655779607184
// MI455X (gfx1250) — compile-verified, matrix operand order checked by reading, not run on hardware
//
#include <hip/hip_runtime.h>
#include <hip/hip_bf16.h>
#include <math.h>

typedef __attribute__((ext_vector_type(16))) __bf16 v16bf;
typedef __attribute__((ext_vector_type(8)))  float  v8f;

#define GN_EPS     1e-5f
#define NEG_SLOPE  0.2f

// ---------- helpers ----------
__device__ __forceinline__ unsigned enc_f32(float f) {
  unsigned u = __float_as_uint(f);
  return (u & 0x80000000u) ? ~u : (u | 0x80000000u);   // monotone map: uint order == float order
}
__device__ __forceinline__ float dec_f32(unsigned u) {
  u = (u & 0x80000000u) ? (u ^ 0x80000000u) : ~u;
  return __uint_as_float(u);
}
__device__ __forceinline__ float elu_f(float x) {
  return x > 0.f ? x : (__expf(x) - 1.f);
}
__device__ __forceinline__ void edge_sd(const int* __restrict__ ei, long long idx,
                                        long long E, int& s, int& d) {
  if (idx < E) { s = ei[idx]; d = ei[E + idx]; }
  else         { s = d = (int)(idx - E); }             // implicit self-loop
}

// ---------- fp32 -> bf16 convert ----------
__global__ void cvt_bf16_kernel(const float* __restrict__ in,
                                __hip_bfloat16* __restrict__ out, long long n) {
  long long t = (long long)blockIdx.x * blockDim.x + threadIdx.x;
  if (t < n) out[t] = __float2bfloat16(in[t]);
}

// ---------- GroupNorm(groups of 8 ch) + ELU, emit bf16 ----------
__global__ void gn_elu_bf16_kernel(const float* __restrict__ in,
                                   const float* __restrict__ w,
                                   const float* __restrict__ b,
                                   __hip_bfloat16* __restrict__ out,
                                   long long N, int C) {
  int G = C >> 3;                                      // group size fixed = 8
  long long t = (long long)blockIdx.x * blockDim.x + threadIdx.x;
  if (t >= N * (long long)G) return;
  long long n = t / G;
  int g = (int)(t % G);
  const float* p = in + n * C + g * 8;
  float v[8], mu = 0.f;
#pragma unroll
  for (int i = 0; i < 8; ++i) { v[i] = p[i]; mu += v[i]; }
  mu *= 0.125f;
  float var = 0.f;
#pragma unroll
  for (int i = 0; i < 8; ++i) { float d0 = v[i] - mu; var += d0 * d0; }
  var *= 0.125f;
  float inv = rsqrtf(var + GN_EPS);
  __hip_bfloat16* o = out + n * C + g * 8;
#pragma unroll
  for (int i = 0; i < 8; ++i) {
    int c = g * 8 + i;
    float y = (v[i] - mu) * inv * w[c] + b[c];
    o[i] = __float2bfloat16(elu_f(y));
  }
}

// ---------- bf16 WMMA GEMM, 2x2 register-blocked (32x32 out per wave) ----------
// out[M,Nc] = A[M,K] @ W[Nc,K]^T (+bias) (+= out if accumulate)
// block = 256 threads = 8 waves arranged 2(rows) x 4(cols): 64 x 128 out per block.
template <int K>
__global__ void gemm_bf16_wmma_kernel(const __hip_bfloat16* __restrict__ A,
                                      const __hip_bfloat16* __restrict__ W,
                                      const float* __restrict__ bias,
                                      float* __restrict__ out,
                                      int M, int Nc, int accumulate) {
  int wave = threadIdx.x >> 5, lane = threadIdx.x & 31;
  int row0 = blockIdx.x * 64 + (wave >> 2) * 32;
  int col0 = blockIdx.y * 128 + (wave & 3) * 32;
  if (row0 + 32 > M || col0 + 32 > Nc) return;
  int mn = lane & 15;                // A row within tile / B column within tile
  int kb = (lane >> 4) << 3;         // half-wave K offset (0/8); also M/N offset in C/D
  const unsigned* Ar0 = (const unsigned*)(A + (size_t)(row0 + mn) * K);
  const unsigned* Ar1 = (const unsigned*)(A + (size_t)(row0 + 16 + mn) * K);
  const unsigned* Wr0 = (const unsigned*)(W + (size_t)(col0 + mn) * K);
  const unsigned* Wr1 = (const unsigned*)(W + (size_t)(col0 + 16 + mn) * K);
  union Frag { unsigned u[8]; v16bf v; };
  v8f acc00 = {}, acc01 = {}, acc10 = {}, acc11 = {};
#pragma unroll
  for (int k0 = 0; k0 < K; k0 += 32) {
    Frag fa0, fa1, fb0, fb1;
#pragma unroll
    for (int c = 0; c < 8; ++c) {
      int kh = ((c < 4) ? (2 * c) : (2 * c + 8)) + kb;   // ISA 16-bit A-layout K pairs
      int di = (k0 + kh) >> 1;                           // dword index (bf16 pair)
      fa0.u[c] = Ar0[di]; fa1.u[c] = Ar1[di];
      fb0.u[c] = Wr0[di]; fb1.u[c] = Wr1[di];
    }
    acc00 = __builtin_amdgcn_wmma_f32_16x16x32_bf16(false, fa0.v, false, fb0.v, (short)0, acc00, false, false);
    acc01 = __builtin_amdgcn_wmma_f32_16x16x32_bf16(false, fa0.v, false, fb1.v, (short)0, acc01, false, false);
    acc10 = __builtin_amdgcn_wmma_f32_16x16x32_bf16(false, fa1.v, false, fb0.v, (short)0, acc10, false, false);
    acc11 = __builtin_amdgcn_wmma_f32_16x16x32_bf16(false, fa1.v, false, fb1.v, (short)0, acc11, false, false);
  }
  // epilogue: C/D layout -> vgpr r holds M=r (lanes 0-15) / M=r+8 (lanes 16-31)
  v8f accs[2][2] = {{acc00, acc01}, {acc10, acc11}};
#pragma unroll
  for (int ti = 0; ti < 2; ++ti) {
#pragma unroll
    for (int tj = 0; tj < 2; ++tj) {
      int col = col0 + tj * 16 + mn;
      float bv = bias ? bias[col] : 0.f;
#pragma unroll
      for (int r = 0; r < 8; ++r) {
        int row = row0 + ti * 16 + r + kb;
        float v = accs[ti][tj][r] + bv;
        size_t o = (size_t)row * Nc + col;
        if (accumulate) v += out[o];
        out[o] = v;
      }
    }
  }
}

// ---------- alpha_src / alpha_dst matvecs (wave per node, C = 256) ----------
__global__ void attn_matvec_kernel(const float* __restrict__ h,
                                   const float* __restrict__ aw_s,
                                   const float* __restrict__ aw_d,
                                   float* __restrict__ asrc, float* __restrict__ adst,
                                   long long N, int C) {
  int wave = threadIdx.x >> 5, lane = threadIdx.x & 31;
  long long n = (long long)blockIdx.x * (blockDim.x >> 5) + wave;
  if (n >= N) return;
  int c0 = lane * 8;
  const float* hp = h + n * (long long)C + c0;
  float s0 = 0.f, s1 = 0.f;
#pragma unroll
  for (int i = 0; i < 8; ++i) {
    float x = hp[i];
    s0 += x * aw_s[c0 + i];
    s1 += x * aw_d[c0 + i];
  }
  for (int off = 16; off; off >>= 1) {
    s0 += __shfl_xor(s0, off, 32);
    s1 += __shfl_xor(s1, off, 32);
  }
  if (lane == 0) { asrc[n] = s0; adst[n] = s1; }
}

// ---------- segment-softmax init ----------
__global__ void init_seg_kernel(unsigned* __restrict__ mkey, float* __restrict__ ssum,
                                long long N) {
  long long t = (long long)blockIdx.x * blockDim.x + threadIdx.x;
  if (t < N) { mkey[t] = 0x007FFFFFu /* enc(-inf) */; ssum[t] = 0.f; }
}
__global__ void init_bias_rows_kernel(float* __restrict__ out,
                                      const float* __restrict__ bias,
                                      long long N, int C) {
  long long t = (long long)blockIdx.x * blockDim.x + threadIdx.x;
  if (t < N * (long long)C) out[t] = bias[t % C];
}

// ---------- edge phase 1: logits + segment max ----------
__global__ void edge_logits_kernel(const int* __restrict__ ei,
                                   const float* __restrict__ asrc,
                                   const float* __restrict__ adst,
                                   float* __restrict__ e, unsigned* __restrict__ mkey,
                                   long long E, long long N) {
  long long idx = (long long)blockIdx.x * blockDim.x + threadIdx.x;
  if (idx >= E + N) return;
  int s, d; edge_sd(ei, idx, E, s, d);
  float v = asrc[s] + adst[d];
  v = v > 0.f ? v : NEG_SLOPE * v;
  e[idx] = v;
  atomicMax(&mkey[d], enc_f32(v));
}

// ---------- edge phase 2: exp + segment sum ----------
__global__ void edge_expsum_kernel(const int* __restrict__ ei,
                                   float* __restrict__ e,
                                   const unsigned* __restrict__ mkey,
                                   float* __restrict__ ssum,
                                   long long E, long long N) {
  long long idx = (long long)blockIdx.x * blockDim.x + threadIdx.x;
  if (idx >= E + N) return;
  int s, d; edge_sd(ei, idx, E, s, d);
  float ex = __expf(e[idx] - dec_f32(mkey[d]));
  e[idx] = ex;
  atomicAdd(&ssum[d], ex);
}

// ---------- edge phase 3: weighted scatter-add (wave per edge) ----------
__global__ void edge_aggregate_kernel(const int* __restrict__ ei,
                                      const float* __restrict__ h,
                                      const float* __restrict__ e,
                                      const float* __restrict__ ssum,
                                      float* __restrict__ out,
                                      long long E, long long N, int C) {
  int wave = threadIdx.x >> 5, lane = threadIdx.x & 31;
  long long idx = (long long)blockIdx.x * (blockDim.x >> 5) + wave;
  if (idx >= E + N) return;
  int s, d; edge_sd(ei, idx, E, s, d);
  float alpha = e[idx] / (ssum[d] + 1e-16f);
  int c0 = lane * 8;
  const float4* hp = (const float4*)(h + (size_t)s * C + c0);
  float4 a = hp[0], b = hp[1];
  float* op = out + (size_t)d * C + c0;
  atomicAdd(op + 0, a.x * alpha); atomicAdd(op + 1, a.y * alpha);
  atomicAdd(op + 2, a.z * alpha); atomicAdd(op + 3, a.w * alpha);
  atomicAdd(op + 4, b.x * alpha); atomicAdd(op + 5, b.y * alpha);
  atomicAdd(op + 6, b.z * alpha); atomicAdd(op + 7, b.w * alpha);
}

// ---------- host ----------
static inline int cdivi(long long a, long long b) { return (int)((a + b - 1) / b); }

extern "C" void kernel_launch(void* const* d_in, const int* in_sizes, int n_in,
                              void* d_out, int out_size, void* d_ws, size_t ws_size,
                              hipStream_t stream) {
  const float* x       = (const float*)d_in[0];
  const int*   ei      = (const int*)  d_in[1];
  const float* pre_w   = (const float*)d_in[2];
  const float* pre_b   = (const float*)d_in[3];
  const float* lin1_w  = (const float*)d_in[4];
  const float* lin1_b  = (const float*)d_in[5];
  const float* n1_w    = (const float*)d_in[6];
  const float* n1_b    = (const float*)d_in[7];
  const float* g1_w    = (const float*)d_in[8];
  const float* g1_as   = (const float*)d_in[9];
  const float* g1_ad   = (const float*)d_in[10];
  const float* g1_b    = (const float*)d_in[11];
  const float* n2_w    = (const float*)d_in[12];
  const float* n2_b    = (const float*)d_in[13];
  const float* g2_w    = (const float*)d_in[14];
  const float* g2_as   = (const float*)d_in[15];
  const float* g2_ad   = (const float*)d_in[16];
  const float* g2_b    = (const float*)d_in[17];
  const float* n3_w    = (const float*)d_in[18];
  const float* n3_b    = (const float*)d_in[19];
  const float* lin2_w  = (const float*)d_in[20];
  const float* lin2_b  = (const float*)d_in[21];
  const float* skip_w  = (const float*)d_in[22];
  const float* skip_b  = (const float*)d_in[23];

  const int CIN = 256, H = 256, COUT = 512;
  const long long N = in_sizes[0] / CIN;
  const long long E = in_sizes[1] / 2;
  const long long ET = E + N;
  float* out = (float*)d_out;

  // carve workspace
  char* p = (char*)d_ws;
  auto carve = [&](size_t bytes) -> char* {
    char* r = p; p += (bytes + 255) & ~(size_t)255; return r;
  };
  __hip_bfloat16* xb   = (__hip_bfloat16*)carve((size_t)N * CIN * 2);
  __hip_bfloat16* actb = (__hip_bfloat16*)carve((size_t)N * H * 2);
  __hip_bfloat16* w1b  = (__hip_bfloat16*)carve((size_t)H * CIN * 2);
  __hip_bfloat16* g1wb = (__hip_bfloat16*)carve((size_t)H * H * 2);
  __hip_bfloat16* g2wb = (__hip_bfloat16*)carve((size_t)H * H * 2);
  __hip_bfloat16* w2b  = (__hip_bfloat16*)carve((size_t)COUT * H * 2);
  __hip_bfloat16* skb  = (__hip_bfloat16*)carve((size_t)COUT * CIN * 2);
  float*    F1   = (float*)   carve((size_t)N * H * 4);
  float*    F2   = (float*)   carve((size_t)N * H * 4);
  float*    ebuf = (float*)   carve((size_t)ET * 4);
  unsigned* mkey = (unsigned*)carve((size_t)N * 4);
  float*    ssum = (float*)   carve((size_t)N * 4);
  float*    asrc = (float*)   carve((size_t)N * 4);
  float*    adst = (float*)   carve((size_t)N * 4);

  const int TB = 256;
  // weight + input conversions to bf16
  cvt_bf16_kernel<<<cdivi((long long)H * CIN, TB), TB, 0, stream>>>(lin1_w, w1b, (long long)H * CIN);
  cvt_bf16_kernel<<<cdivi((long long)H * H,   TB), TB, 0, stream>>>(g1_w,  g1wb, (long long)H * H);
  cvt_bf16_kernel<<<cdivi((long long)H * H,   TB), TB, 0, stream>>>(g2_w,  g2wb, (long long)H * H);
  cvt_bf16_kernel<<<cdivi((long long)COUT * H,  TB), TB, 0, stream>>>(lin2_w, w2b, (long long)COUT * H);
  cvt_bf16_kernel<<<cdivi((long long)COUT * CIN, TB), TB, 0, stream>>>(skip_w, skb, (long long)COUT * CIN);
  cvt_bf16_kernel<<<cdivi(N * CIN, TB), TB, 0, stream>>>(x, xb, N * CIN);

  // GEMM grids: block = 64 rows x 128 cols (8 waves, each 32x32 via 2x2 WMMA tiles)
  dim3 gH((int)(N / 64), H / 128);
  dim3 gO((int)(N / 64), COUT / 128);

  // pre_norm + elu -> lin1
  gn_elu_bf16_kernel<<<cdivi(N * (CIN / 8), TB), TB, 0, stream>>>(x, pre_w, pre_b, actb, N, CIN);
  gemm_bf16_wmma_kernel<256><<<gH, 256, 0, stream>>>(actb, w1b, lin1_b, F1, (int)N, H, 0);

  // ---- GAT layer 1 ----
  gn_elu_bf16_kernel<<<cdivi(N * (H / 8), TB), TB, 0, stream>>>(F1, n1_w, n1_b, actb, N, H);
  gemm_bf16_wmma_kernel<256><<<gH, 256, 0, stream>>>(actb, g1wb, (const float*)nullptr, F1, (int)N, H, 0);
  attn_matvec_kernel<<<cdivi(N, 8), 256, 0, stream>>>(F1, g1_as, g1_ad, asrc, adst, N, H);
  init_seg_kernel<<<cdivi(N, TB), TB, 0, stream>>>(mkey, ssum, N);
  init_bias_rows_kernel<<<cdivi(N * H, TB), TB, 0, stream>>>(F2, g1_b, N, H);
  edge_logits_kernel<<<cdivi(ET, TB), TB, 0, stream>>>(ei, asrc, adst, ebuf, mkey, E, N);
  edge_expsum_kernel<<<cdivi(ET, TB), TB, 0, stream>>>(ei, ebuf, mkey, ssum, E, N);
  edge_aggregate_kernel<<<cdivi(ET, 8), 256, 0, stream>>>(ei, F1, ebuf, ssum, F2, E, N, H);

  // ---- GAT layer 2 ----
  gn_elu_bf16_kernel<<<cdivi(N * (H / 8), TB), TB, 0, stream>>>(F2, n2_w, n2_b, actb, N, H);
  gemm_bf16_wmma_kernel<256><<<gH, 256, 0, stream>>>(actb, g2wb, (const float*)nullptr, F1, (int)N, H, 0);
  attn_matvec_kernel<<<cdivi(N, 8), 256, 0, stream>>>(F1, g2_as, g2_ad, asrc, adst, N, H);
  init_seg_kernel<<<cdivi(N, TB), TB, 0, stream>>>(mkey, ssum, N);
  init_bias_rows_kernel<<<cdivi(N * H, TB), TB, 0, stream>>>(F2, g2_b, N, H);
  edge_logits_kernel<<<cdivi(ET, TB), TB, 0, stream>>>(ei, asrc, adst, ebuf, mkey, E, N);
  edge_expsum_kernel<<<cdivi(ET, TB), TB, 0, stream>>>(ei, ebuf, mkey, ssum, E, N);
  edge_aggregate_kernel<<<cdivi(ET, 8), 256, 0, stream>>>(ei, F1, ebuf, ssum, F2, E, N, H);

  // ---- output: lin2 then fused residual skip GEMM accumulating into d_out ----
  gn_elu_bf16_kernel<<<cdivi(N * (H / 8), TB), TB, 0, stream>>>(F2, n3_w, n3_b, actb, N, H);
  gemm_bf16_wmma_kernel<256><<<gO, 256, 0, stream>>>(actb, w2b, lin2_b, out, (int)N, COUT, 0);
  gemm_bf16_wmma_kernel<256><<<gO, 256, 0, stream>>>(xb, skb, skip_b, out, (int)N, COUT, 1);
}
